// RNN_12498354831511
// MI455X (gfx1250) — matrix operand order repaired and checked by reading, not compiled, not run on hardware
//
#include <hip/hip_runtime.h>
#include <hip/hip_bf16.h>

// ---------------------------------------------------------------------------
// Problem constants (from reference): T=8192, D_INP=D_HID=D_OUT=2048
// ---------------------------------------------------------------------------
#define TT      8192
#define DD      2048
#define STEPS   (TT - 1)          // 8191 recurrence steps / GEMM rows
#define KDIM    2048
#define NTILES  (DD / 16)         // 128 n-tiles
#define KTILES  (KDIM / 32)       // 64 k-steps
#define REC_BLOCKS 128            // 128 blocks * 16 waves = 2048 waves (one per row)
#define REC_THREADS 512

typedef __attribute__((ext_vector_type(16))) __bf16     v16bf;
typedef __attribute__((ext_vector_type(8)))  float      v8f;
typedef __attribute__((ext_vector_type(4)))  unsigned   v4u;

union Frag { v16bf v; v4u u[2]; };

static __device__ __forceinline__ __bf16 f2bf(float f) {
    union { float f; unsigned u; } a; a.f = f;
    unsigned r = a.u + 0x7FFFu + ((a.u >> 16) & 1u);   // round-to-nearest-even
    unsigned short h = (unsigned short)(r >> 16);
    __bf16 out;
    __builtin_memcpy(&out, &h, 2);
    return out;
}

static __device__ __forceinline__ float sigmoidf_fast(float x) {
    return 1.0f / (1.0f + __expf(-x));
}

// ---------------------------------------------------------------------------
// init: copy output row 0 = caption[0], reset grid-barrier state
// ---------------------------------------------------------------------------
__global__ void init_kernel(const float* __restrict__ cap,
                            float* __restrict__ out,
                            unsigned* __restrict__ bar) {
    int g = blockIdx.x * blockDim.x + threadIdx.x;
    if (g < DD) out[g] = cap[g];
    if (g == 0) { bar[0] = 0u; bar[1] = 0u; }
}

// ---------------------------------------------------------------------------
// fp32 -> bf16 elementwise (caption rows 0..T-2)
// ---------------------------------------------------------------------------
__global__ void cvt_bf16_kernel(const float* __restrict__ src,
                                __bf16* __restrict__ dst, int n) {
    int g = blockIdx.x * blockDim.x + threadIdx.x;
    if (g < n) dst[g] = f2bf(src[g]);
}

// ---------------------------------------------------------------------------
// Pack a [2048 x K-slice] weight matrix (row-major, row = output n, col = k)
// into WMMA B-fragment order: dst[((kt*128 + nt)*32 + lane)*16 + i]
//   lane 0-15 : K = kt*32 + i      , N = nt*16 + lane
//   lane16-31 : K = kt*32 + 16 + i , N = nt*16 + lane-16
// ---------------------------------------------------------------------------
__global__ void pack_B_kernel(const float* __restrict__ W,
                              __bf16* __restrict__ dst,
                              int ld, int coloff) {
    int e = blockIdx.x * blockDim.x + threadIdx.x;
    if (e >= DD * KDIM) return;
    int i  = e & 15;
    int L  = (e >> 4) & 31;
    int nt = (e >> 9) & (NTILES - 1);
    int kt = e >> 16;
    int k  = kt * 32 + (L >> 4) * 16 + i;
    int n  = nt * 16 + (L & 15);
    dst[e] = f2bf(W[(size_t)n * ld + coloff + k]);
}

// ---------------------------------------------------------------------------
// WMMA GEMM:  C[M x 2048] = A[M x 2048](bf16, row-major) * Bpacked + bias
// One wave -> 16(M) x 64(N); block = 8 waves (128 rows), grid = (32, 64).
// Double-buffered fragments: loads for k-step kt+1 are issued before the
// four back-to-back WMMAs of k-step kt, so v_wmma overlaps vmem latency.
// SIGMOID epilogue used for the output projection.
// ---------------------------------------------------------------------------
template <bool SIGMOID>
__global__ void __launch_bounds__(256)
gemm_bf16_kernel(const __bf16* __restrict__ A,
                 const __bf16* __restrict__ Bp,
                 const float* __restrict__ bias,
                 float* __restrict__ C,
                 int M, int rowOff) {
    const int wave = threadIdx.x >> 5;
    const int lane = threadIdx.x & 31;
    const int kh   = lane >> 4;
    const int ln   = lane & 15;

    const int mtile = blockIdx.y * 8 + wave;
    const int mbase = mtile * 16;
    const int nt0   = blockIdx.x * 4;          // 4 n-tiles = 64 columns

    int arow = mbase + ln;
    if (arow >= M) arow = M - 1;               // clamp (row 15 of last tile unused)
    const __bf16* aptr = A + (size_t)arow * KDIM + kh * 16;
    // base B-fragment pointer for kt=0 (nt0 and lane folded in);
    // per n-tile stride = 512 elems, per k-step stride = NTILES*512 elems
    const __bf16* bptr = Bp + (((size_t)nt0 * 32) + lane) * 16;

    v8f acc[4] = {v8f{}, v8f{}, v8f{}, v8f{}};

    Frag a, b[4];
    a.u[0] = *(const v4u*)(aptr);
    a.u[1] = *(const v4u*)(aptr + 8);
#pragma unroll
    for (int t = 0; t < 4; ++t) {
        b[t].u[0] = *(const v4u*)(bptr + t * 512);
        b[t].u[1] = *(const v4u*)(bptr + t * 512 + 8);
    }

    for (int kt = 0; kt < KTILES - 1; ++kt) {
        // issue next k-step's loads first (one big clause, loads in flight
        // while the current WMMAs execute)
        const __bf16* ap2 = aptr + (size_t)(kt + 1) * 32;
        const __bf16* bp2 = bptr + (size_t)(kt + 1) * (NTILES * 512);
        Frag an, bn[4];
        an.u[0] = *(const v4u*)(ap2);
        an.u[1] = *(const v4u*)(ap2 + 8);
#pragma unroll
        for (int t = 0; t < 4; ++t) {
            bn[t].u[0] = *(const v4u*)(bp2 + t * 512);
            bn[t].u[1] = *(const v4u*)(bp2 + t * 512 + 8);
        }
#pragma unroll
        for (int t = 0; t < 4; ++t) {
            acc[t] = __builtin_amdgcn_wmma_f32_16x16x32_bf16(
                false, a.v, false, b[t].v, (short)0, acc[t], false, false);
        }
        a = an;
#pragma unroll
        for (int t = 0; t < 4; ++t) b[t] = bn[t];
    }
#pragma unroll
    for (int t = 0; t < 4; ++t) {
        acc[t] = __builtin_amdgcn_wmma_f32_16x16x32_bf16(
            false, a.v, false, b[t].v, (short)0, acc[t], false, false);
    }

#pragma unroll
    for (int v = 0; v < 8; ++v) {
        int row = mbase + kh * 8 + v;
        if (row >= M) continue;
#pragma unroll
        for (int t = 0; t < 4; ++t) {
            int col = (nt0 + t) * 16 + ln;
            float val = acc[t][v] + bias[col];
            if (SIGMOID) val = sigmoidf_fast(val);
            C[(size_t)(row + rowOff) * DD + col] = val;
        }
    }
}

// ---------------------------------------------------------------------------
// Device-wide sense barrier (counters in global scratch, init'd each launch)
// ---------------------------------------------------------------------------
static __device__ __forceinline__ void grid_barrier(unsigned* bar, unsigned nblk) {
    __syncthreads();
    if (threadIdx.x == 0) {
        __threadfence();
        unsigned gen = __hip_atomic_load(&bar[1], __ATOMIC_ACQUIRE,
                                         __HIP_MEMORY_SCOPE_AGENT);
        unsigned arrived = atomicAdd(&bar[0], 1u);
        if (arrived == nblk - 1u) {
            atomicExch(&bar[0], 0u);
            __threadfence();
            atomicAdd(&bar[1], 1u);
        } else {
            while (__hip_atomic_load(&bar[1], __ATOMIC_ACQUIRE,
                                     __HIP_MEMORY_SCOPE_AGENT) == gen)
                __builtin_amdgcn_s_sleep(1);
        }
    }
    __syncthreads();
}

// ---------------------------------------------------------------------------
// Persistent recurrence: h_{t+1} = sigmoid(A[t] + W_hh * h_t)
// 2048 waves, one per output row. W_hh row lives in 64 VGPRs per lane
// (loaded once). Per step: 64 FMAs vs broadcast h, wave32 reduction,
// v_exp sigmoid, device barrier.
// ---------------------------------------------------------------------------
__global__ void __launch_bounds__(REC_THREADS, 1)
rnn_recurrence_kernel(const float* __restrict__ Wh,     // [2048 x 4096]
                      const float* __restrict__ Aarr,   // [STEPS x 2048]
                      const float* __restrict__ feat,   // [2048]
                      float* __restrict__ hbuf,         // [2 x 2048]
                      __bf16* __restrict__ Hbf,         // [STEPS x 2048]
                      unsigned* __restrict__ bar) {
    const int lane = threadIdx.x & 31;
    const int wave = threadIdx.x >> 5;
    const int row  = blockIdx.x * (REC_THREADS / 32) + wave;   // 0..2047

    // W_hh = W_h[:, 2048:4096]; lane handles k = lane + 32*i
    float w[64];
    const float* wp = Wh + (size_t)row * 4096 + 2048 + lane;
#pragma unroll
    for (int i = 0; i < 64; ++i) w[i] = wp[i * 32];

    // h0 = features
    int gid = blockIdx.x * blockDim.x + threadIdx.x;
    if (gid < DD) hbuf[gid] = feat[gid];
    __threadfence();
    grid_barrier(bar, gridDim.x);

    for (int t = 0; t < STEPS; ++t) {
        const float* hc = hbuf + (size_t)(t & 1) * DD;
        float s = 0.f;
#pragma unroll
        for (int i = 0; i < 64; ++i)
            s = fmaf(w[i], hc[lane + 32 * i], s);
#pragma unroll
        for (int off = 16; off > 0; off >>= 1)
            s += __shfl_down(s, off, 32);
        if (lane == 0) {
            float hn = sigmoidf_fast(s + Aarr[(size_t)t * DD + row]);
            hbuf[(size_t)((t & 1) ^ 1) * DD + row] = hn;
            Hbf[(size_t)t * DD + row] = f2bf(hn);
        }
        __threadfence();
        grid_barrier(bar, gridDim.x);
    }
}

// ---------------------------------------------------------------------------
// Host-side launch sequence
// ---------------------------------------------------------------------------
extern "C" void kernel_launch(void* const* d_in, const int* in_sizes, int n_in,
                              void* d_out, int out_size, void* d_ws, size_t ws_size,
                              hipStream_t stream) {
    const float* features = (const float*)d_in[0];   // [2048]
    const float* caption  = (const float*)d_in[1];   // [8192 x 2048]
    const float* W_h      = (const float*)d_in[2];   // [2048 x 4096]
    const float* b_h      = (const float*)d_in[3];   // [2048]
    const float* W_o      = (const float*)d_in[4];   // [2048 x 2048]
    const float* b_o      = (const float*)d_in[5];   // [2048]
    float* out            = (float*)d_out;           // [8192 x 2048]

    // ---- workspace carve-up (256B aligned) ----
    char* ws = (char*)d_ws;
    size_t off = 0;
    unsigned* bar  = (unsigned*)(ws + off);             off += 256;
    float*    hbuf = (float*)(ws + off);                off += (size_t)2 * DD * 4;      // 16640
    __bf16*   Xbf  = (__bf16*)(ws + off);               // [STEPS x 2048] bf16
    size_t xbytes  = (size_t)STEPS * DD * 2;            off += xbytes;
    __bf16*   Hbf  = Xbf;                               // reuse: X dead after phase 1
    __bf16*   WxP  = (__bf16*)(ws + off);               off += (size_t)DD * KDIM * 2;
    __bf16*   WoP  = (__bf16*)(ws + off);               off += (size_t)DD * KDIM * 2;
    float*    Aarr = (float*)(ws + off);                off += (size_t)STEPS * DD * 4;
    (void)ws_size; (void)in_sizes; (void)n_in; (void)out_size;

    // 1) out row 0 = caption[0]; reset barrier
    init_kernel<<<(DD + 255) / 256, 256, 0, stream>>>(caption, out, bar);

    // 2) caption[0..T-2] -> bf16
    {
        int n = STEPS * DD;
        cvt_bf16_kernel<<<(n + 255) / 256, 256, 0, stream>>>(caption, Xbf, n);
    }

    // 3) pack W_x (W_h cols 0..2047) and W_o into WMMA fragment order
    {
        int n = DD * KDIM;
        pack_B_kernel<<<(n + 255) / 256, 256, 0, stream>>>(W_h, WxP, 4096, 0);
        pack_B_kernel<<<(n + 255) / 256, 256, 0, stream>>>(W_o, WoP, 2048, 0);
    }

    // 4) A = X * W_x^T + b_h   (bf16 WMMA, fp32 accumulate)
    gemm_bf16_kernel<false><<<dim3(NTILES / 4, 64), 256, 0, stream>>>(
        Xbf, WxP, b_h, Aarr, STEPS, 0);

    // 5) persistent recurrence (register-resident W_hh, device barriers)
    rnn_recurrence_kernel<<<REC_BLOCKS, REC_THREADS, 0, stream>>>(
        W_h, Aarr, features, hbuf, Hbf, bar);

    // 6) Y = sigmoid(H * W_o^T + b_o) -> out rows 1..T-1
    gemm_bf16_kernel<true><<<dim3(NTILES / 4, 64), 256, 0, stream>>>(
        Hbf, WoP, b_o, out, STEPS, 1);
}
